// SKVMN_3487513444396
// MI455X (gfx1250) — matrix operand order repaired and checked by reading, not compiled, not run on hardware
//
#include <hip/hip_runtime.h>
#include <stdint.h>

#define DEV __device__ __forceinline__

typedef __attribute__((ext_vector_type(16))) _Float16 v16h;
typedef __attribute__((ext_vector_type(8)))  float    v8f;
typedef __attribute__((ext_vector_type(4)))  unsigned u32x4;
typedef __attribute__((ext_vector_type(8)))  int      i32x8;
typedef __attribute__((ext_vector_type(4)))  int      i32x4;

// Problem constants
// B=64, T=200, M=50, D=128, NUM_C=4000

// ---------------- workspace layout (bytes) ----------------
#define WS_KF16   0u                       // [B*T,128] f16  (3,276,800)
#define WS_YF16   (WS_KF16 + 3276800u)     // [B*T,128] f16
#define WS_CORR   (WS_YF16 + 3276800u)     // [B*T,64]  f32 (cols 50..63 unused)
#define WS_IVP    (WS_CORR + 3276800u)     // [B*T] uint4 packed 2-bit iv
#define WS_PREV   (WS_IVP  + 204800u)      // [T,B] int
#define WS_MATCH  (WS_PREV + 51200u)       // [T,B] int
#define WS_FT16   (WS_MATCH + 51200u)      // [T,B,128] f16
#define WS_MK16   (WS_FT16 + 3276800u)     // [64,128] f16 (Mk padded)
#define WS_WQ     (WS_MK16 + 16384u)       // fW16|aW16|eW16|adW16 concat (196,608 B)
#define WS_WCAT   (WS_WQ   + 196608u)      // [512,256] f16 (Wih|Whh)
#define WS_BSUM   (WS_WCAT + 262144u)      // [512] f32 bih+bhh
#define WS_MV     (WS_BSUM + 2048u)        // [B,50,128] f32 (1,638,400)
#define WS_HH     (WS_MV   + 1638400u)     // [T,B,128] f32 (6,553,600)
#define WS_HC     (WS_HH   + 6553600u)     // [T,B,128] f32

// ---------------- WMMA fragment helpers (ISA 7.12.2 layouts) ----------------
// A: 16x32 f16, row m = lane&15, lane-half selects K groups {0..7,16..23} / {8..15,24..31}
DEV v16h load_a_frag(const _Float16* A, int lda, int row0, int k0, int lane) {
  int m = lane & 15, hf = lane >> 4;
  const _Float16* rp = A + (size_t)(row0 + m) * lda + k0 + 8 * hf;
  v16h a;
#pragma unroll
  for (int v = 0; v < 8; ++v) {
    int kb = ((v & 4) ? 16 : 0) + 2 * (v & 3);
    a[2 * v]     = rp[kb];
    a[2 * v + 1] = rp[kb + 1];
  }
  return a;
}

// B: 32x16 where B[k][n] = W[n][k] (W row-major [N][K]); lane n = lane&15 reads
// W[n0+n][k0 + 8*(lane>>4) + {0..7,16..23}] (same K groups as A).
DEV v16h load_b_frag(const _Float16* W, int ldw, int n0, int k0, int lane) {
  int n = lane & 15, koff = (lane >> 4) * 8;
  const _Float16* rp = W + (size_t)(n0 + n) * ldw + k0 + koff;
  v16h b;
#pragma unroll
  for (int i = 0; i < 16; ++i) b[i] = rp[(i < 8) ? i : i + 8];
  return b;
}

DEV v8f wmma_f16(v16h a, v16h b, v8f c) {
  return __builtin_amdgcn_wmma_f32_16x16x32_f16(false, a, false, b, (short)0, c,
                                                false, false);
}

DEV float sigf(float x) { return 1.0f / (1.0f + expf(-x)); }

// ---------------- Tensor Data Mover: 1D global->LDS copy ----------------
// data_size=3 (8B units); n8 = number of 8-byte units (<= 32768 fits tile_dim0).
// 6-arg builtin variant (clang-23 / therock-10.0 headers).
DEV void tdm_load_1d(const void* gp, unsigned lds_off, unsigned n8) {
  unsigned long long ga = (unsigned long long)(size_t)gp;
  u32x4 g0 = { 1u,                                   // count=1, gather off
               lds_off,                              // lds_addr
               (unsigned)ga,                         // global_addr[31:0]
               (unsigned)((ga >> 32) & 0x1FFFFFFu) | (2u << 30) }; // [56:32] | type=2
  i32x8 g1 = { (int)(3u << 16),                      // data_size=8B, mask=0
               (int)((n8 & 0xFFFFu) << 16),          // tensor_dim0 lo16
               (int)((n8 >> 16) | (1u << 16)),       // tensor_dim0 hi16 | tensor_dim1=1
               (int)((n8 & 0xFFFFu) << 16),          // tile_dim0
               0,                                    // tile_dim1/2 unused
               (int)n8,                              // tensor_dim0_stride lo32
               0, 0 };
  i32x4 g2 = {0, 0, 0, 0}, g3 = {0, 0, 0, 0};
  i32x8 g4 = {0, 0, 0, 0, 0, 0, 0, 0};
  __builtin_amdgcn_tensor_load_to_lds(g0, g1, g2, g3, g4, 0);
}

// ---------------- K0: weight conversion / packing ----------------
__global__ void prep_kernel(const float* fW, const float* aW, const float* eW,
                            const float* adW, const float* Wih, const float* Whh,
                            const float* bih, const float* bhh, const float* Mk,
                            _Float16* wq, _Float16* wcat, _Float16* mk16,
                            float* bsum) {
  int i = blockIdx.x * 256 + threadIdx.x;           // 0 .. 131071
  if (i < 32768) { wq[i] = (_Float16)fW[i]; wq[32768 + i] = (_Float16)aW[i]; }
  if (i < 16384) { wq[65536 + i] = (_Float16)eW[i]; wq[81920 + i] = (_Float16)adW[i]; }
  if (i < 131072) {
    int o = i >> 8, k = i & 255;
    wcat[i] = (_Float16)((k < 128) ? Wih[o * 128 + k] : Whh[o * 128 + (k - 128)]);
  }
  if (i < 8192) { int r_ = i >> 7; mk16[i] = (r_ < 50) ? (_Float16)Mk[i] : (_Float16)0.0f; }
  if (i < 512) bsum[i] = bih[i] + bhh[i];
}

// ---------------- K1a: embedding gathers -> f16 ----------------
__global__ void embed_kernel(const int* q, const int* r, const float* k_emb,
                             const float* x_emb, _Float16* kf16, _Float16* yf16) {
  int i = blockIdx.x * 256 + threadIdx.x;           // 0 .. 1,638,399
  int row = i >> 7, d = i & 127;
  int qq = q[row], rr = r[row];
  kf16[i] = (_Float16)k_emb[(size_t)qq * 128 + d];
  yf16[i] = (_Float16)x_emb[(size_t)(qq + 4000 * rr) * 128 + d];
}

// ---------------- K1b: corr = softmax(k @ Mk^T), iv pack ----------------
__global__ void corr_kernel(const _Float16* kf16, const _Float16* mk16,
                            float* corr, uint4* ivp) {
  __shared__ float logits[128 * 64];
  int lane = threadIdx.x & 31, wave = threadIdx.x >> 5;
  int row0 = blockIdx.x * 128;
  {
    v8f z = {0, 0, 0, 0, 0, 0, 0, 0};
    v8f acc[4];
#pragma unroll
    for (int nt = 0; nt < 4; ++nt) acc[nt] = z;
#pragma unroll
    for (int kc = 0; kc < 4; ++kc) {
      v16h a = load_a_frag(kf16, 128, row0 + wave * 16, kc * 32, lane);
#pragma unroll
      for (int nt = 0; nt < 4; ++nt) {
        v16h b = load_b_frag(mk16, 128, nt * 16, kc * 32, lane);
        acc[nt] = wmma_f16(a, b, acc[nt]);
      }
    }
    int n = lane & 15, lh = lane >> 4;
#pragma unroll
    for (int nt = 0; nt < 4; ++nt)
#pragma unroll
      for (int vi = 0; vi < 8; ++vi)
        logits[(wave * 16 + vi + 8 * lh) * 64 + nt * 16 + n] = acc[nt][vi];
  }
  __syncthreads();
  if (threadIdx.x < 128) {
    int r_ = threadIdx.x;
    int grow = row0 + r_;
    float mx = -1e30f;
    for (int m = 0; m < 50; ++m) mx = fmaxf(mx, logits[r_ * 64 + m]);
    float s = 0.0f;
    for (int m = 0; m < 50; ++m) s += expf(logits[r_ * 64 + m] - mx);
    float inv = 1.0f / s;
    unsigned w0 = 0, w1 = 0, w2 = 0, w3 = 0;
    for (int m = 0; m < 50; ++m) {
      float p = expf(logits[r_ * 64 + m] - mx) * inv;
      corr[(size_t)grow * 64 + m] = p;
      float tri = fmaxf(fminf((p - 0.075f) / 0.013f, (1.0f - p) / 0.912f), 0.0f);
      unsigned ivv = (tri >= 0.6f) ? 2u : ((tri >= 0.1f) ? 1u : 0u);
      unsigned sh = (unsigned)((m & 15) * 2);
      if (m < 16) w0 |= ivv << sh;
      else if (m < 32) w1 |= ivv << sh;
      else if (m < 48) w2 |= ivv << sh;
      else w3 |= ivv << sh;
    }
    ivp[grow] = make_uint4(w0, w1, w2, w3);
  }
}

// ---------------- K2: most-recent identical step ----------------
__global__ void prevmatch_kernel(const uint4* ivp, int* prevp, int* matchp) {
  int gid = blockIdx.x * 256 + threadIdx.x;
  if (gid >= 64 * 200) return;
  int b = gid / 200, i = gid % 200;
  uint4 me = ivp[b * 200 + i];
  int pj = 0, mt = 0;
  for (int j = i - 1; j >= 0; --j) {
    uint4 o = ivp[b * 200 + j];
    if (o.x == me.x && o.y == me.y && o.z == me.z && o.w == me.w) { pj = j; mt = 1; break; }
  }
  prevp[i * 64 + b] = pj;
  matchp[i * 64 + b] = mt;
}

// ---------------- K3: DKVMN memory scan (16 batches / WG) ----------------
__global__ void memscan_kernel(const float* corr, const _Float16* kf16,
                               const _Float16* yf16, const _Float16* wq_g,
                               const float* Mv0, const float* f_b, const float* a_b,
                               const float* e_b, const float* ad_b, float* mv,
                               _Float16* ft16) {
  extern __shared__ char smem[];
  _Float16* FW  = (_Float16*)smem;        // 128x256
  _Float16* AW  = FW + 32768;             // 128x256
  _Float16* EW  = AW + 32768;             // 128x128
  _Float16* ADW = EW + 16384;             // 128x128
  _Float16* A1  = ADW + 16384;            // 16x256  [read | k]
  _Float16* A2  = A1 + 4096;              // 16x256  [f | y]
  _Float16* WE  = A2 + 4096;              // 16x128
  float*    EST = (float*)(WE + 2048);    // 16x128
  float*    ADST= EST + 2048;             // 16x128
  float*    CT  = ADST + 2048;            // 16x64

  int tid = threadIdx.x, lane = tid & 31, wave = tid >> 5;
  int b0 = blockIdx.x * 16;

  if (wave == 0) {                         // TDM preload of all 4 weight mats
    tdm_load_1d(wq_g, (unsigned)(size_t)FW, 196608u / 8u);
    __builtin_amdgcn_s_wait_tensorcnt(0);
  }
  for (int e = 0; e < 400; ++e) {          // mv init from Mv0 broadcast
    int idx = tid + e * 256;
    int bb = idx / 6400, rem = idx % 6400;
    mv[((size_t)(b0 + bb) * 50) * 128 + rem] = Mv0[rem];
  }
  __syncthreads();

  int n = lane & 15, lh = lane >> 4;
  for (int t = 0; t < 200; ++t) {
    for (int e = 0; e < 4; ++e) {          // stage c_t
      int idx = tid + e * 256;
      int bb = idx >> 6, m = idx & 63;
      CT[idx] = (m < 50) ? corr[(((size_t)(b0 + bb) * 200 + t) * 64) + m] : 0.0f;
    }
    for (int e = 0; e < 8; ++e) {          // stage k_t, y_t halves of A1/A2
      int idx = tid + e * 256;
      int bb = idx >> 7, d = idx & 127;
      size_t ro = ((size_t)(b0 + bb) * 200 + t) * 128 + d;
      A1[bb * 256 + 128 + d] = kf16[ro];
      A2[bb * 256 + 128 + d] = yf16[ro];
    }
    __syncthreads();
    for (int e = 0; e < 8; ++e) {          // read = c_t @ mv
      int idx = tid + e * 256;
      int bb = idx >> 7, d = idx & 127;
      float s = 0.0f;
      const float* mrow = mv + ((size_t)(b0 + bb) * 50) * 128 + d;
      for (int m = 0; m < 50; ++m) s += CT[bb * 64 + m] * mrow[m * 128];
      A1[bb * 256 + d] = (_Float16)s;
    }
    __syncthreads();
    {                                      // GEMM1: f = tanh(A1 @ fW^T + f_b)
      v8f acc = {0, 0, 0, 0, 0, 0, 0, 0};
#pragma unroll
      for (int kc = 0; kc < 8; ++kc) {
        v16h a = load_a_frag(A1, 256, 0, kc * 32, lane);
        v16h b = load_b_frag(FW, 256, wave * 16, kc * 32, lane);
        acc = wmma_f16(a, b, acc);
      }
      int d = wave * 16 + n;
      float bias = f_b[d];
#pragma unroll
      for (int vi = 0; vi < 8; ++vi) {
        int bb = vi + 8 * lh;
        float v = tanhf(acc[vi] + bias);
        A2[bb * 256 + d] = (_Float16)v;
        ft16[((size_t)t * 64 + b0 + bb) * 128 + d] = (_Float16)v;
      }
    }
    __syncthreads();
    {                                      // GEMM2: we = A2 @ aW^T + a_b
      v8f acc = {0, 0, 0, 0, 0, 0, 0, 0};
#pragma unroll
      for (int kc = 0; kc < 8; ++kc) {
        v16h a = load_a_frag(A2, 256, 0, kc * 32, lane);
        v16h b = load_b_frag(AW, 256, wave * 16, kc * 32, lane);
        acc = wmma_f16(a, b, acc);
      }
      int d = wave * 16 + n;
      float bias = a_b[d];
#pragma unroll
      for (int vi = 0; vi < 8; ++vi)
        WE[(vi + 8 * lh) * 128 + d] = (_Float16)(acc[vi] + bias);
    }
    __syncthreads();
    {                                      // GEMM3/4: e = sig(..), ad = tanh(..)
      v8f accE = {0, 0, 0, 0, 0, 0, 0, 0};
      v8f accA = {0, 0, 0, 0, 0, 0, 0, 0};
#pragma unroll
      for (int kc = 0; kc < 4; ++kc) {
        v16h a  = load_a_frag(WE, 128, 0, kc * 32, lane);
        v16h be = load_b_frag(EW, 128, wave * 16, kc * 32, lane);
        v16h ba = load_b_frag(ADW, 128, wave * 16, kc * 32, lane);
        accE = wmma_f16(a, be, accE);
        accA = wmma_f16(a, ba, accA);
      }
      int d = wave * 16 + n;
      float be_ = e_b[d], ba_ = ad_b[d];
#pragma unroll
      for (int vi = 0; vi < 8; ++vi) {
        int bb = vi + 8 * lh;
        EST[bb * 128 + d]  = sigf(accE[vi] + be_);
        ADST[bb * 128 + d] = tanhf(accA[vi] + ba_);
      }
    }
    __syncthreads();
    for (int e = 0; e < 400; ++e) {        // mv = mv*(1 - c*e) + c*ad
      int idx = tid + e * 256;
      int bb = idx / 6400, rem = idx % 6400;
      int m = rem >> 7, d = rem & 127;
      float c = CT[bb * 64 + m];
      size_t off = ((size_t)(b0 + bb) * 50 + m) * 128 + d;
      mv[off] = mv[off] * (1.0f - c * EST[bb * 128 + d]) + c * ADST[bb * 128 + d];
    }
    __syncthreads();
  }
}

// ---------------- K4: hop-LSTM, persistent single WG ----------------
__global__ void lstm_kernel(const _Float16* ft16, const _Float16* wcat_g,
                            const float* bsum, const int* prevp, const int* matchp,
                            const float* hx0, const float* cx0, const float* pW,
                            const float* pb, float* Hh, float* Hc, float* out) {
  extern __shared__ char smem[];
  _Float16* WC = (_Float16*)smem;          // 512x256 [Wih|Whh] f16
  _Float16* HX = WC + 131072;              // 64x128 f16
  float*    CX = (float*)(HX + 8192);      // 64x128 f32

  int tid = threadIdx.x, lane = tid & 31, wave = tid >> 5;
  if (wave == 0) {
    tdm_load_1d(wcat_g, (unsigned)(size_t)WC, 262144u / 8u);
    __builtin_amdgcn_s_wait_tensorcnt(0);
  }
  for (int e = 0; e < 8; ++e) {
    int idx = tid + e * 1024;
    int d = idx & 127;
    HX[idx] = (_Float16)hx0[d];
    CX[idx] = cx0[d];
  }
  __syncthreads();

  int mt = wave >> 3, ntb = wave & 7, n = lane & 15, lh = lane >> 4;
  float bias[4];
#pragma unroll
  for (int g = 0; g < 4; ++g) bias[g] = bsum[g * 128 + ntb * 16 + n];

  for (int t = 0; t < 200; ++t) {
    for (int e = 0; e < 8; ++e) {          // hop: jump to last identical step
      int idx = tid + e * 1024;
      int b = idx >> 7, d = idx & 127;
      if (matchp[t * 64 + b]) {
        int pv = prevp[t * 64 + b];
        size_t off = ((size_t)pv * 64 + b) * 128 + d;
        HX[idx] = (_Float16)Hh[off];
        CX[idx] = Hc[off];
      }
    }
    __syncthreads();

    v8f z = {0, 0, 0, 0, 0, 0, 0, 0};
    v8f acc[4];
#pragma unroll
    for (int g = 0; g < 4; ++g) acc[g] = z;
    const _Float16* ftT = ft16 + (size_t)t * 64 * 128;
#pragma unroll
    for (int kc = 0; kc < 8; ++kc) {
      v16h a = (kc < 4) ? load_a_frag(ftT, 128, mt * 16, kc * 32, lane)
                        : load_a_frag(HX, 128, mt * 16, (kc - 4) * 32, lane);
#pragma unroll
      for (int g = 0; g < 4; ++g) {        // i,f,g,o gate tiles for same dims
        v16h b = load_b_frag(WC, 256, g * 128 + ntb * 16, kc * 32, lane);
        acc[g] = wmma_f16(a, b, acc[g]);
      }
    }
    __syncthreads();                        // all HX reads done before rewrite

#pragma unroll
    for (int vi = 0; vi < 8; ++vi) {
      int b = mt * 16 + vi + 8 * lh, d = ntb * 16 + n;
      float ig = sigf(acc[0][vi] + bias[0]);
      float fg = sigf(acc[1][vi] + bias[1]);
      float gg = tanhf(acc[2][vi] + bias[2]);
      float og = sigf(acc[3][vi] + bias[3]);
      float c1 = fg * CX[b * 128 + d] + ig * gg;
      float h1 = og * tanhf(c1);
      CX[b * 128 + d] = c1;
      HX[b * 128 + d] = (_Float16)h1;
      size_t off = ((size_t)t * 64 + b) * 128 + d;
      Hh[off] = h1;
      Hc[off] = c1;
    }
    __syncthreads();

    if (tid < 64) {                         // p = sigmoid(hx @ p_W + p_b)
      float s = pb[0];
      for (int k = 0; k < 128; ++k) s += (float)HX[tid * 128 + k] * pW[k];
      out[tid * 200 + t] = sigf(s);
    }
    __syncthreads();
  }
}

// ---------------- host launcher ----------------
extern "C" void kernel_launch(void* const* d_in, const int* in_sizes, int n_in,
                              void* d_out, int out_size, void* d_ws, size_t ws_size,
                              hipStream_t stream) {
  const int*   q       = (const int*)d_in[0];
  const int*   r       = (const int*)d_in[1];
  const float* Mk      = (const float*)d_in[2];
  const float* Mv0     = (const float*)d_in[3];
  const float* k_emb   = (const float*)d_in[4];
  const float* x_emb   = (const float*)d_in[5];
  const float* f_W     = (const float*)d_in[6];
  const float* f_b     = (const float*)d_in[7];
  const float* a_W     = (const float*)d_in[8];
  const float* a_b     = (const float*)d_in[9];
  const float* erase_W = (const float*)d_in[10];
  const float* erase_b = (const float*)d_in[11];
  const float* add_W   = (const float*)d_in[12];
  const float* add_b   = (const float*)d_in[13];
  const float* Wih     = (const float*)d_in[14];
  const float* Whh     = (const float*)d_in[15];
  const float* bih     = (const float*)d_in[16];
  const float* bhh     = (const float*)d_in[17];
  const float* hx0     = (const float*)d_in[18];
  const float* cx0     = (const float*)d_in[19];
  const float* pW      = (const float*)d_in[20];
  const float* pb      = (const float*)d_in[21];

  char* ws = (char*)d_ws;
  _Float16* kf16  = (_Float16*)(ws + WS_KF16);
  _Float16* yf16  = (_Float16*)(ws + WS_YF16);
  float*    corr  = (float*)(ws + WS_CORR);
  uint4*    ivp   = (uint4*)(ws + WS_IVP);
  int*      prevp = (int*)(ws + WS_PREV);
  int*      matchp= (int*)(ws + WS_MATCH);
  _Float16* ft16  = (_Float16*)(ws + WS_FT16);
  _Float16* mk16  = (_Float16*)(ws + WS_MK16);
  _Float16* wq    = (_Float16*)(ws + WS_WQ);
  _Float16* wcat  = (_Float16*)(ws + WS_WCAT);
  float*    bsum  = (float*)(ws + WS_BSUM);
  float*    mv    = (float*)(ws + WS_MV);
  float*    Hh    = (float*)(ws + WS_HH);
  float*    Hc    = (float*)(ws + WS_HC);
  float*    out   = (float*)d_out;

  prep_kernel<<<512, 256, 0, stream>>>(f_W, a_W, erase_W, add_W, Wih, Whh, bih,
                                       bhh, Mk, wq, wcat, mk16, bsum);
  embed_kernel<<<6400, 256, 0, stream>>>(q, r, k_emb, x_emb, kf16, yf16);
  corr_kernel<<<100, 256, 0, stream>>>(kf16, mk16, corr, ivp);
  prevmatch_kernel<<<50, 256, 0, stream>>>(ivp, prevp, matchp);
  memscan_kernel<<<4, 256, 237568, stream>>>(corr, kf16, yf16, wq, Mv0, f_b, a_b,
                                             erase_b, add_b, mv, ft16);
  lstm_kernel<<<1, 1024, 311296, stream>>>(ft16, wcat, bsum, prevp, matchp, hx0,
                                           cx0, pW, pb, Hh, Hc, out);
}
